// MixerGroupedTiedAttention_50843822850199
// MI455X (gfx1250) — matrix operand order repaired and checked by reading, not compiled, not run on hardware
//
#include <hip/hip_runtime.h>
#include <hip/hip_bf16.h>

// ---------------------------------------------------------------------------
// MixerGroupedTiedAttention for MI455X (gfx1250, wave32, WMMA bf16 16x16x32)
//
// Pipeline:
//   0) convert/transpose: x -> bf16 ; W_qkv/W_rk/W_g -> bf16 transposed (N,K)
//   1) gemm_wmma_bf16:  async-to-LDS double-buffered bf16 WMMA GEMM
//   2) postprocess:     RMSNorm + RoPE + logpos scaling -> bf16 Q/K/V^T
//   3) attn:            windowed causal flash attention, async K/V staging
//   4) gate_silu:       out = y * silu(g)
// ---------------------------------------------------------------------------

typedef __attribute__((ext_vector_type(16))) __bf16        v16bf;
typedef __attribute__((ext_vector_type(8)))  float         v8f;
typedef __attribute__((ext_vector_type(4)))  unsigned int  u32x4;

union FragU {
  u32x4 u[2];
  v16bf v;
};

__device__ __forceinline__ unsigned short f2bf(float f) {
  unsigned int u = __builtin_bit_cast(unsigned int, f);
  u += 0x7FFFu + ((u >> 16) & 1u);        // round-to-nearest-even
  return (unsigned short)(u >> 16);
}
__device__ __forceinline__ unsigned int pack2bf(float lo, float hi) {
  return (unsigned int)f2bf(lo) | ((unsigned int)f2bf(hi) << 16);
}

// 32-bit LDS byte offset of a __shared__ object (addrspace(3) view)
__device__ __forceinline__ unsigned lds_off(const void* p) {
  return (unsigned)(unsigned long long)(const __attribute__((address_space(3))) char*)p;
}
// CDNA5 async DMA: global -> LDS, 16 bytes per lane, tracked by ASYNCcnt
__device__ __forceinline__ void async_copy16(unsigned lds_byte_off, const void* gptr) {
  asm volatile("global_load_async_to_lds_b128 %0, %1, off"
               :: "v"(lds_byte_off), "v"(gptr) : "memory");
}
__device__ __forceinline__ void wait_async0() {
  asm volatile("s_wait_asynccnt 0" ::: "memory");
}

// A-fragment (16x32 bf16, MxK): lane<16 -> M=lane, K 0..7 & 16..23
//                               lane>=16 -> M=lane-16, K 8..15 & 24..31
__device__ __forceinline__ v16bf load_afrag(const unsigned short* p, int ld) {
  const int lane = threadIdx.x & 31;
  const int m = lane & 15;
  const int ko = (lane & 16) ? 8 : 0;
  FragU r;
  r.u[0] = *(const u32x4*)(p + m * ld + ko);
  r.u[1] = *(const u32x4*)(p + m * ld + ko + 16);
  return r.v;
}
// B-fragment (32x16 bf16, KxN) from K-contiguous [n][k] storage:
// lane<16 -> N=lane, K 0..7 & 16..23 ; lane>=16 -> N=lane-16, K 8..15 & 24..31
__device__ __forceinline__ v16bf load_bfrag(const unsigned short* p, int ld) {
  const int lane = threadIdx.x & 31;
  const int n = lane & 15;
  const int ko = (lane & 16) ? 8 : 0;
  FragU r;
  r.u[0] = *(const u32x4*)(p + n * ld + ko);
  r.u[1] = *(const u32x4*)(p + n * ld + ko + 16);
  return r.v;
}

__device__ __forceinline__ float wave_sum32(float x) {
  #pragma unroll
  for (int o = 16; o > 0; o >>= 1) x += __shfl_xor(x, o, 32);
  return x;
}
__device__ __forceinline__ float half_max16(float x) {
  #pragma unroll
  for (int o = 8; o > 0; o >>= 1) x = fmaxf(x, __shfl_xor(x, o, 16));
  return x;
}
__device__ __forceinline__ float half_sum16(float x) {
  #pragma unroll
  for (int o = 8; o > 0; o >>= 1) x += __shfl_xor(x, o, 16);
  return x;
}

// ---------------------------------------------------------------------------
// Kernel 0a: fp32 -> bf16, 8 elements/thread
// ---------------------------------------------------------------------------
__global__ __launch_bounds__(256)
void convert_bf16(const float* __restrict__ in, unsigned short* __restrict__ outb,
                  int n8) {
  const int i = blockIdx.x * blockDim.x + threadIdx.x;
  if (i < n8) {
    const float4 a = ((const float4*)in)[2 * i];
    const float4 b = ((const float4*)in)[2 * i + 1];
    u32x4 o = { pack2bf(a.x, a.y), pack2bf(a.z, a.w),
                pack2bf(b.x, b.y), pack2bf(b.z, b.w) };
    ((u32x4*)outb)[i] = o;
  }
}

// ---------------------------------------------------------------------------
// Kernel 0b: fp32 (K,N) -> bf16 (N,K) tiled transpose (dims multiples of 32)
// ---------------------------------------------------------------------------
__global__ __launch_bounds__(256)
void transpose_bf16(const float* __restrict__ in, unsigned short* __restrict__ outb,
                    int K, int N) {
  __shared__ unsigned short tile[32][33];
  const int k0 = blockIdx.y * 32, n0 = blockIdx.x * 32;
  const int tx = threadIdx.x & 31, ty = threadIdx.x >> 5;   // ty: 0..7
  #pragma unroll
  for (int i = 0; i < 4; ++i)
    tile[ty + 8 * i][tx] = f2bf(in[(size_t)(k0 + ty + 8 * i) * N + n0 + tx]);
  __syncthreads();
  #pragma unroll
  for (int i = 0; i < 4; ++i)
    outb[(size_t)(n0 + ty + 8 * i) * K + k0 + tx] = tile[tx][ty + 8 * i];
}

// ---------------------------------------------------------------------------
// Kernel 1: C[M,N] = A[M,K] @ Bt[N,K]^T (+bias), bf16 in, fp32 out.
// 128x128 tile per block, 8 waves, K step 32, async-to-LDS double buffered.
// ---------------------------------------------------------------------------
__global__ __launch_bounds__(256, 2)
void gemm_wmma_bf16(const unsigned short* __restrict__ A,   // M x K bf16
                    const unsigned short* __restrict__ Bt,  // N x K bf16
                    const float* __restrict__ bias, float* __restrict__ C,
                    int M, int N, int K) {
  constexpr int TM = 128, TN = 128, TK = 32;
  __shared__ __align__(16) unsigned short Asl[2][TM * TK];  // [m][k]
  __shared__ __align__(16) unsigned short Bsl[2][TN * TK];  // [n][k]
  const int tid  = threadIdx.x;
  const int lane = tid & 31;
  const int wave = tid >> 5;
  const int bm = blockIdx.y * TM;
  const int bn = blockIdx.x * TN;
  const int wm = (wave & 3) * 32;
  const int wn = (wave >> 2) * 64;

  const v8f vzero = {};
  v8f acc[2][4];
  #pragma unroll
  for (int i = 0; i < 2; ++i)
    #pragma unroll
    for (int j = 0; j < 4; ++j) acc[i][j] = vzero;

  // async stage of one 128x32 bf16 tile from A and one from Bt (pure copies)
  auto stage = [&](int buf, int kk) {
    #pragma unroll
    for (int s = 0; s < 2; ++s) {
      const int c   = tid + 256 * s;      // 0..511 16B chunks
      const int row = c >> 2;             // 0..127
      const int k8  = (c & 3) * 8;        // bf16 offset in tile row
      async_copy16(lds_off(&Asl[buf][row * TK + k8]),
                   A + (size_t)(bm + row) * K + kk + k8);
      async_copy16(lds_off(&Bsl[buf][row * TK + k8]),
                   Bt + (size_t)(bn + row) * K + kk + k8);
    }
  };

  stage(0, 0);
  int cur = 0;
  for (int kk = 0; kk < K; kk += TK) {
    wait_async0();       // current tile landed in LDS (per-wave)
    __syncthreads();     // ... for every wave
    if (kk + TK < K) stage(cur ^ 1, kk + TK);   // overlap next tile with WMMA

    const unsigned short* As = Asl[cur];
    const unsigned short* Bs = Bsl[cur];
    const v16bf af0 = load_afrag(&As[(wm)      * TK], TK);
    const v16bf af1 = load_afrag(&As[(wm + 16) * TK], TK);
    #pragma unroll
    for (int j = 0; j < 4; ++j) {
      const v16bf bf = load_bfrag(&Bs[(wn + 16 * j) * TK], TK);
      acc[0][j] = __builtin_amdgcn_wmma_f32_16x16x32_bf16(
          false, af0, false, bf, (short)0, acc[0][j], false, false);
      acc[1][j] = __builtin_amdgcn_wmma_f32_16x16x32_bf16(
          false, af1, false, bf, (short)0, acc[1][j], false, false);
    }
    cur ^= 1;
  }
  // epilogue: VGPR r holds M = r + 8*(lane>=16), N = lane&15
  const int hi = lane >> 4;
  const int cN = lane & 15;
  #pragma unroll
  for (int j = 0; j < 4; ++j) {
    const int col = bn + wn + 16 * j + cN;
    if (col < N) {
      const float bv = bias ? bias[col] : 0.f;
      #pragma unroll
      for (int i = 0; i < 2; ++i) {
        const int row0 = bm + wm + 16 * i + 8 * hi;
        #pragma unroll
        for (int r = 0; r < 8; ++r)
          C[(size_t)(row0 + r) * N + col] = acc[i][j][r] + bv;
      }
    }
  }
}

// ---------------------------------------------------------------------------
// Kernel 2: per-(b,t) RMSNorm + RoPE + logpos scale; writes bf16 Q/K/V^T.
//   Q: (B,16,T,128)  K: (B,4,T,128)  V^T: (B,4,128,T)
// ---------------------------------------------------------------------------
__global__ __launch_bounds__(256)
void postprocess(const float* __restrict__ qkv, const float* __restrict__ krope,
                 const float* __restrict__ scaler,
                 unsigned short* __restrict__ Qb, unsigned short* __restrict__ Kb,
                 unsigned short* __restrict__ Vtb, int T) {
  const int bt   = blockIdx.x;
  const int b    = bt / T;
  const int t    = bt - b * T;
  const int lane = threadIdx.x & 31;
  const int wave = threadIdx.x >> 5;     // 8 waves
  const float* base = qkv + (size_t)bt * (20 * 128);
  const float logpos  = __logf(fminf((float)(t + 1), 1024.f));
  const float invfreq = __powf(10000.f, -(float)lane * (1.f / 32.f));
  const float ang = (float)t * invfreq;
  const float cs = __cosf(ang), sn = __sinf(ang);

  #pragma unroll
  for (int hh = 0; hh < 2; ++hh) {
    const int h = wave + 8 * hh;
    const float* v = base + h * 128;
    const float x0 = v[lane], x1 = v[lane + 32], x2 = v[lane + 64], x3 = v[lane + 96];
    const float ss = wave_sum32(x0 * x0 + x1 * x1 + x2 * x2 + x3 * x3);
    const float rms = __frsqrt_rn(ss * (1.f / 128.f) + 1e-6f);
    const float sc = scaler[h] * logpos * rms;
    unsigned short* q = Qb + ((size_t)(b * 16 + h) * T + t) * 128;
    q[lane]      = f2bf(sc * x0);
    q[lane + 32] = f2bf(sc * x1);
    q[lane + 64] = f2bf(sc * ( x2 * cs + x3 * sn));
    q[lane + 96] = f2bf(sc * (-x2 * sn + x3 * cs));
  }
  if (wave < 4) {
    const int g = wave;
    const float* v = base + (16 + g) * 128;
    const float x0 = v[lane], x1 = v[lane + 32], x2 = v[lane + 64], x3 = v[lane + 96];
    const float ss = wave_sum32(x0 * x0 + x1 * x1 + x2 * x2 + x3 * x3);
    const float rms = __frsqrt_rn(ss * (1.f / 128.f) + 1e-6f);
    const float kt0 = rms * x0, kt1 = rms * x1;   // kv_tied
    const float vv0 = rms * x2, vv1 = rms * x3;   // v
    const float* kr = krope + (size_t)bt * 64;    // not rmsnormed
    const float k1 = kr[lane], k2 = kr[lane + 32];
    const float kr0 =  k1 * cs + k2 * sn;
    const float kr1 = -k1 * sn + k2 * cs;
    unsigned short* kd = Kb + ((size_t)(b * 4 + g) * T + t) * 128;
    kd[lane]      = f2bf(kt0);
    kd[lane + 32] = f2bf(kt1);
    kd[lane + 64] = f2bf(kr0);
    kd[lane + 96] = f2bf(kr1);
    unsigned short* vt = Vtb + (size_t)(b * 4 + g) * 128 * T + t;
    vt[(size_t)(lane)      * T] = f2bf(kt0);
    vt[(size_t)(lane + 32) * T] = f2bf(kt1);
    vt[(size_t)(lane + 64) * T] = f2bf(vv0);
    vt[(size_t)(lane + 96) * T] = f2bf(vv1);
  }
}

// ---------------------------------------------------------------------------
// Kernel 3: windowed causal attention, flash-style, all-WMMA,
// async-to-LDS double-buffered K/V tiles.
// Block: 128 threads (4 waves), 64 query rows, 32-key tiles.
// ---------------------------------------------------------------------------
__global__ __launch_bounds__(128, 2)
void attn(const unsigned short* __restrict__ Qb, const unsigned short* __restrict__ Kb,
          const unsigned short* __restrict__ Vtb, float* __restrict__ out, int T) {
  constexpr int QT = 64, KT = 32;
  __shared__ __align__(16) unsigned short Ksl[2][KT * 128];   // [key][d]
  __shared__ __align__(16) unsigned short Vsl[2][128 * KT];   // [d][key]
  __shared__ __align__(16) unsigned short Psl[4][16 * KT];    // per-wave P
  const int tid = threadIdx.x, lane = tid & 31, wave = tid >> 5;
  const int nqt = T / QT;
  const int qt  = blockIdx.x % nqt;
  const int bh  = blockIdx.x / nqt;
  const int h   = bh & 15;
  const int b   = bh >> 4;
  const int g   = h >> 2;
  const int q0  = qt * QT;
  const int qbase = q0 + wave * 16;

  const unsigned short* qptr = Qb  + (size_t)(b * 16 + h) * T * 128;
  const unsigned short* kptr = Kb  + (size_t)(b * 4 + g) * T * 128;
  const unsigned short* vptr = Vtb + (size_t)(b * 4 + g) * 128 * T;

  FragU qf[4];
  {
    const int m  = lane & 15;
    const int ko = (lane & 16) ? 8 : 0;
    const unsigned short* qrow = qptr + (size_t)(qbase + m) * 128;
    #pragma unroll
    for (int f = 0; f < 4; ++f) {
      qf[f].u[0] = *(const u32x4*)(qrow + 32 * f + ko);
      qf[f].u[1] = *(const u32x4*)(qrow + 32 * f + ko + 16);
    }
  }

  const v8f vzero = {};
  v8f acc[8];
  #pragma unroll
  for (int c = 0; c < 8; ++c) acc[c] = vzero;
  float rmax[8], rsum[8];
  #pragma unroll
  for (int r = 0; r < 8; ++r) { rmax[r] = -1e30f; rsum[r] = 0.f; }

  int js = q0 - 1024;
  if (js < 0) js = 0;
  js &= ~(KT - 1);
  const int je = q0 + QT;
  const float sscale = 0.08838834764831845f;   // 1/sqrt(128)
  const int hi = lane >> 4;
  const int cN = lane & 15;

  // async stage of one 32-key K tile and V^T tile (pure bf16 copies)
  auto stageKV = [&](int buf, int jt) {
    #pragma unroll
    for (int s = 0; s < 4; ++s) {
      const int idx = tid + 128 * s;          // 0..511 16B chunks
      const int key = idx >> 4;
      const int d8  = (idx & 15) * 8;
      async_copy16(lds_off(&Ksl[buf][key * 128 + d8]),
                   kptr + (size_t)(jt + key) * 128 + d8);
    }
    #pragma unroll
    for (int s = 0; s < 4; ++s) {
      const int idx = tid + 128 * s;
      const int d  = idx >> 2;
      const int k8 = (idx & 3) * 8;
      async_copy16(lds_off(&Vsl[buf][d * KT + k8]),
                   vptr + (size_t)d * T + jt + k8);
    }
  };

  stageKV(0, js);
  int cur = 0;
  for (int jt = js; jt < je; jt += KT) {
    wait_async0();
    __syncthreads();
    if (jt + KT < je) stageKV(cur ^ 1, jt + KT);   // overlap with compute

    const unsigned short* Ks = Ksl[cur];
    const unsigned short* Vs = Vsl[cur];

    // ---- S = Q K^T (two 16-key subtiles) ----
    v8f s[2];
    #pragma unroll
    for (int sub = 0; sub < 2; ++sub) {
      v8f sv = vzero;
      #pragma unroll
      for (int f = 0; f < 4; ++f) {
        const v16bf kf = load_bfrag(&Ks[(16 * sub) * 128 + 32 * f], 128);
        sv = __builtin_amdgcn_wmma_f32_16x16x32_bf16(
            false, qf[f].v, false, kf, (short)0, sv, false, false);
      }
      s[sub] = sv;
    }

    // ---- online softmax (per-row across 16 lanes of each half-wave) ----
    #pragma unroll
    for (int r = 0; r < 8; ++r) {
      const int row = qbase + r + 8 * hi;
      const int j0 = jt + cN, j1 = jt + 16 + cN;
      float v0 = s[0][r] * sscale;
      float v1 = s[1][r] * sscale;
      v0 = (j0 <= row && j0 >= row - 1024) ? v0 : -1e30f;
      v1 = (j1 <= row && j1 >= row - 1024) ? v1 : -1e30f;
      const float tmax = half_max16(fmaxf(v0, v1));
      const float nm = fmaxf(rmax[r], tmax);
      const float a = __expf(rmax[r] - nm);
      const float p0 = (v0 > -1e29f) ? __expf(v0 - nm) : 0.f;
      const float p1 = (v1 > -1e29f) ? __expf(v1 - nm) : 0.f;
      rsum[r] = rsum[r] * a + half_sum16(p0 + p1);
      rmax[r] = nm;
      #pragma unroll
      for (int c = 0; c < 8; ++c) acc[c][r] *= a;
      Psl[wave][(r + 8 * hi) * KT + cN]      = f2bf(p0);
      Psl[wave][(r + 8 * hi) * KT + 16 + cN] = f2bf(p1);
    }

    // ---- O += P V (P re-layout via per-wave LDS; same-wave LDS in-order) ----
    const v16bf pf = load_afrag(&Psl[wave][0], KT);
    #pragma unroll
    for (int c = 0; c < 8; ++c) {
      const v16bf vf = load_bfrag(&Vs[(16 * c) * KT], KT);
      acc[c] = __builtin_amdgcn_wmma_f32_16x16x32_bf16(
          false, pf, false, vf, (short)0, acc[c], false, false);
    }
    cur ^= 1;
  }

  // ---- normalize and store y as (B,T,H,128) ----
  #pragma unroll
  for (int r = 0; r < 8; ++r) {
    const float inv = 1.f / rsum[r];
    const int row = qbase + r + 8 * hi;
    float* op = out + ((size_t)(b * T + row) * 16 + h) * 128;
    #pragma unroll
    for (int c = 0; c < 8; ++c) op[16 * c + cN] = acc[c][r] * inv;
  }
}

// ---------------------------------------------------------------------------
// Kernel 4: out *= silu(g)
// ---------------------------------------------------------------------------
__global__ __launch_bounds__(256)
void gate_silu(float* __restrict__ out, const float* __restrict__ g, int n4) {
  const int i = blockIdx.x * blockDim.x + threadIdx.x;
  if (i < n4) {
    float4 y = ((const float4*)out)[i];
    const float4 gv = ((const float4*)g)[i];
    y.x *= gv.x / (1.f + __expf(-gv.x));
    y.y *= gv.y / (1.f + __expf(-gv.y));
    y.z *= gv.z / (1.f + __expf(-gv.z));
    y.w *= gv.w / (1.f + __expf(-gv.w));
    ((float4*)out)[i] = y;
  }
}

// ---------------------------------------------------------------------------
extern "C" void kernel_launch(void* const* d_in, const int* in_sizes, int n_in,
                              void* d_out, int out_size, void* d_ws, size_t ws_size,
                              hipStream_t stream) {
  (void)in_sizes; (void)n_in; (void)out_size; (void)ws_size;
  const float* x      = (const float*)d_in[0];   // (B,T,2048)
  const float* Wqkv   = (const float*)d_in[1];   // (2048,2560)
  const float* Wrk    = (const float*)d_in[2];   // (2048,64)
  const float* brk    = (const float*)d_in[3];   // (64,)
  const float* scaler = (const float*)d_in[4];   // (16,)
  const float* Wg     = (const float*)d_in[5];   // (2048,2048)
  float* out = (float*)d_out;

  constexpr int B = 2, T = 2048, DM = 2048;
  constexpr int M = B * T;                       // 4096

  // ---- workspace layout (~138 MB) ----
  unsigned short* xb     = (unsigned short*)d_ws;              // M x 2048
  unsigned short* Wqkv_t = xb + (size_t)M * DM;                // 2560 x 2048
  unsigned short* Wrk_t  = Wqkv_t + (size_t)2560 * DM;         // 128(pad) x 2048
  unsigned short* Wg_t   = Wrk_t + (size_t)128 * DM;           // 2048 x 2048
  unsigned short* Qb     = Wg_t + (size_t)DM * DM;             // B*16*T*128
  unsigned short* Kb     = Qb + (size_t)B * 16 * T * 128;      // B*4*T*128
  unsigned short* Vtb    = Kb + (size_t)B * 4 * T * 128;       // B*4*128*T
  float* qkv_ws   = (float*)(Vtb + (size_t)B * 4 * 128 * T);   // M x 2560
  float* krope_ws = qkv_ws + (size_t)M * 2560;                 // M x 64
  float* g_ws     = krope_ws + (size_t)M * 64;                 // M x 2048

  const dim3 blk256(256);
  // 0) one-time bf16 conversion / weight transposes
  convert_bf16<<<dim3((M * DM / 8 + 255) / 256), blk256, 0, stream>>>(x, xb, M * DM / 8);
  transpose_bf16<<<dim3(2560 / 32, DM / 32), blk256, 0, stream>>>(Wqkv, Wqkv_t, DM, 2560);
  transpose_bf16<<<dim3(64 / 32, DM / 32),   blk256, 0, stream>>>(Wrk,  Wrk_t,  DM, 64);
  transpose_bf16<<<dim3(2048 / 32, DM / 32), blk256, 0, stream>>>(Wg,   Wg_t,   DM, 2048);

  // 1) GEMMs (all bf16 WMMA, async-to-LDS pipelined)
  gemm_wmma_bf16<<<dim3(2560 / 128, M / 128), blk256, 0, stream>>>(
      xb, Wqkv_t, nullptr, qkv_ws, M, 2560, DM);
  gemm_wmma_bf16<<<dim3(1, M / 128), blk256, 0, stream>>>(
      xb, Wrk_t, brk, krope_ws, M, 64, DM);
  gemm_wmma_bf16<<<dim3(2048 / 128, M / 128), blk256, 0, stream>>>(
      xb, Wg_t, nullptr, g_ws, M, DM, DM);

  // 2) norm/rope/scale -> bf16 Q/K/V^T
  postprocess<<<dim3(M), blk256, 0, stream>>>(qkv_ws, krope_ws, scaler, Qb, Kb, Vtb, T);

  // 3) attention
  attn<<<dim3(B * 16 * (T / 64)), dim3(128), 0, stream>>>(Qb, Kb, Vtb, out, T);

  // 4) gating
  const int n4 = M * DM / 4;
  gate_silu<<<dim3((n4 + 255) / 256), blk256, 0, stream>>>(out, g_ws, n4);
}
